// TransferLearningGNN_48369921687560
// MI455X (gfx1250) — matrix operand order repaired and checked by reading, not compiled, not run on hardware
//
#include <hip/hip_runtime.h>
#include <hip/hip_bf16.h>

// ---------------------------------------------------------------------------
// CDNA5 / gfx1250. wave32. bf16 WMMA 16x16x32 for all GEMMs; bf16 operands are
// pre-converted/pre-transposed in global memory so tile staging and fragment
// loads are all 16-byte vector accesses. Tile staging uses the CDNA5 async
// global->LDS path (ASYNCcnt-tracked) — builtin confirmed present.
// ---------------------------------------------------------------------------

typedef __attribute__((ext_vector_type(16))) __bf16 v16bf;
typedef __attribute__((ext_vector_type(8)))  __bf16 v8bf;
typedef __attribute__((ext_vector_type(4)))  __bf16 v4bf;
typedef __attribute__((ext_vector_type(8)))  float  v8f;
typedef int v4i __attribute__((vector_size(4 * sizeof(int))));

#define WMMA_BF16(a, b, c) \
  __builtin_amdgcn_wmma_f32_16x16x32_bf16(false, (a), false, (b), (short)0, (c), false, false)

#if defined(__gfx1250__) && \
    __has_builtin(__builtin_amdgcn_global_load_async_to_lds_b128) && \
    __has_builtin(__builtin_amdgcn_s_wait_asynccnt)
#define HAVE_ASYNC_LDS 1
#else
#define HAVE_ASYNC_LDS 0
#endif

#if HAVE_ASYNC_LDS
typedef __attribute__((address_space(1))) v4i gv4i;
typedef __attribute__((address_space(3))) v4i lv4i;
__device__ __forceinline__ void async_copy16(const void* g, void* l) {
  // AS1 pointer == generic VA; AS3 pointer == low 32 bits of generic LDS VA
  __builtin_amdgcn_global_load_async_to_lds_b128(
      (gv4i*)(uintptr_t)g, (lv4i*)(uintptr_t)l, 0, 0);
}
#endif

__device__ __forceinline__ float elu_f(float x) {
  return x > 0.f ? x : (__expf(x) - 1.f);
}

__device__ __forceinline__ void atomic_add_f32(float* p, float v) {
  __hip_atomic_fetch_add(p, v, __ATOMIC_RELAXED, __HIP_MEMORY_SCOPE_AGENT);
}

// combine two contiguous 16B LDS/global loads into one 16-element fragment
__device__ __forceinline__ v16bf make_frag(const __bf16* lo, const __bf16* hi) {
  v8bf a = *(const v8bf*)lo;
  v8bf b = *(const v8bf*)hi;
  return __builtin_shufflevector(a, b, 0, 1, 2, 3, 4, 5, 6, 7,
                                 8, 9, 10, 11, 12, 13, 14, 15);
}

// ---------------------------------------------------------------------------
// 0a) fp32 -> bf16 streaming convert (8 elements / thread, b128 in, b128 out)
// ---------------------------------------------------------------------------
__global__ __launch_bounds__(256) void cvt_f32_bf16_kernel(
    const float* __restrict__ src, __bf16* __restrict__ dst, int n8) {
  int i = blockIdx.x * 256 + threadIdx.x;
  if (i >= n8) return;
  const float4* s = reinterpret_cast<const float4*>(src) + (size_t)i * 2;
  float4 a = s[0], b = s[1];
  v8bf o;
  o[0] = (__bf16)a.x; o[1] = (__bf16)a.y; o[2] = (__bf16)a.z; o[3] = (__bf16)a.w;
  o[4] = (__bf16)b.x; o[5] = (__bf16)b.y; o[6] = (__bf16)b.z; o[7] = (__bf16)b.w;
  *reinterpret_cast<v8bf*>(dst + (size_t)i * 8) = o;
}

// ---------------------------------------------------------------------------
// 0b) weight prep: W[K][Nc] fp32 -> WT[Nc][K] bf16 (32x32 tiles through LDS)
// ---------------------------------------------------------------------------
__global__ __launch_bounds__(256) void transpose_w_kernel(
    const float* __restrict__ W, __bf16* __restrict__ WT, int K, int Nc) {
  __shared__ float Tl[32][33];
  const int bk = blockIdx.x * 32, bn = blockIdx.y * 32;
  const int t = threadIdx.x;
  const int r = t >> 3, c4 = (t & 7) * 4;
  float4 vv = *reinterpret_cast<const float4*>(&W[(size_t)(bk + r) * Nc + bn + c4]);
  Tl[r][c4] = vv.x; Tl[r][c4 + 1] = vv.y; Tl[r][c4 + 2] = vv.z; Tl[r][c4 + 3] = vv.w;
  __syncthreads();
  v4bf o;
#pragma unroll
  for (int j = 0; j < 4; ++j) o[j] = (__bf16)Tl[c4 + j][r];
  *reinterpret_cast<v4bf*>(&WT[(size_t)(bn + r) * K + bk + c4]) = o;
}

// ---------------------------------------------------------------------------
// 1) Edge gather + segment_sum via fp32 global atomics.
// ---------------------------------------------------------------------------
__global__ __launch_bounds__(256) void scatter_add_kernel(
    const float* __restrict__ x, const int* __restrict__ src,
    const int* __restrict__ dst, const float* __restrict__ ea,
    float* __restrict__ agg, int E, int D, int use_ea) {
  int idx = blockIdx.x * 256 + threadIdx.x;
  int chunks = D >> 2;
  if (idx >= E * chunks) return;
  int e = idx / chunks;
  int c = idx - e * chunks;
  const float4 xv = *reinterpret_cast<const float4*>(&x[(size_t)src[e] * D + c * 4]);
  float s = use_ea ? ea[e] : 1.0f;
  float* a = &agg[(size_t)dst[e] * D + c * 4];
  atomic_add_f32(a + 0, xv.x * s);
  atomic_add_f32(a + 1, xv.y * s);
  atomic_add_f32(a + 2, xv.z * s);
  atomic_add_f32(a + 3, xv.w * s);
}

// ---------------------------------------------------------------------------
// 2) Dual-input GEMM, all-bf16 operands:
//    C[M,Nc] = A1[M,K1]*W1T^T + A2[M,K2]*W2T^T + bias   (WT stored [Nc][K])
//    Block tile 128x128, 8 waves, 32x64 per wave -> 8 WMMAs per k-step.
//    mode: 0 bias; 1 bias+BN+ELU; 2 bias+ELU; 3 bias+residual(fp32)
//    Cf (fp32) and Ch (bf16) output copies are each optional.
// ---------------------------------------------------------------------------
#define GT_M 128
#define GT_N 128
#define GT_K 32
#define LDK  40  // padded LDS K-stride (80 B rows -> 16B aligned, conflict-free)

__global__ __launch_bounds__(256) void gemm_bf16_kernel(
    const __bf16* __restrict__ A1, const __bf16* __restrict__ W1T, int K1,
    const __bf16* __restrict__ A2, const __bf16* __restrict__ W2T, int K2,
    const float* __restrict__ bias, float* __restrict__ Cf,
    __bf16* __restrict__ Ch, int M, int Nc, int mode,
    const float* __restrict__ gamma, const float* __restrict__ beta,
    const float* __restrict__ mean,  const float* __restrict__ var,
    const float* __restrict__ residual) {
  __shared__ __bf16 Alds[GT_M][LDK];   // 128 x 40 bf16 = 10 KB
  __shared__ __bf16 BldsT[GT_N][LDK];  // 128 x 40 bf16 = 10 KB  ([col][k])

  const int t    = threadIdx.x;
  const int lane = t & 31;
  const int wv   = t >> 5;
  const int half = lane >> 4;
  const int l15  = lane & 15;
  const int wrow = (wv & 3) * 32;   // 4 wave-rows
  const int wcol = (wv >> 2) * 64;  // 2 wave-cols
  const int brow = blockIdx.x * GT_M;
  const int bcol = blockIdx.y * GT_N;

  // staging coordinates: 2 threads per row, 16 contiguous bf16 each
  const int sr = t >> 1;
  const int sk = (t & 1) * 16;

  v8f acc[2][4];
#pragma unroll
  for (int i = 0; i < 2; ++i)
#pragma unroll
    for (int j = 0; j < 4; ++j) acc[i][j] = (v8f)0.f;

#pragma unroll 1
  for (int srcid = 0; srcid < 2; ++srcid) {
    const __bf16* A  = srcid ? A2 : A1;
    const __bf16* WT = srcid ? W2T : W1T;
    const int K      = srcid ? K2 : K1;
#pragma unroll 1
    for (int k0 = 0; k0 < K; k0 += GT_K) {
      __syncthreads();
      {
        const __bf16* ap = &A[(size_t)(brow + sr) * K + k0 + sk];
        const __bf16* bp = &WT[(size_t)(bcol + sr) * K + k0 + sk];
#if HAVE_ASYNC_LDS
        // CDNA5 async global->LDS copy: no VGPR round-trip, ASYNCcnt-tracked
        async_copy16(ap,     &Alds[sr][sk]);
        async_copy16(ap + 8, &Alds[sr][sk + 8]);
        async_copy16(bp,     &BldsT[sr][sk]);
        async_copy16(bp + 8, &BldsT[sr][sk + 8]);
        __builtin_amdgcn_s_wait_asynccnt(0);
#else
        // fallback: issue all four 16B loads, then all four LDS stores
        v8bf a0 = *(const v8bf*)ap;
        v8bf a1 = *(const v8bf*)(ap + 8);
        v8bf b0 = *(const v8bf*)bp;
        v8bf b1 = *(const v8bf*)(bp + 8);
        *(v8bf*)&Alds[sr][sk]      = a0;
        *(v8bf*)&Alds[sr][sk + 8]  = a1;
        *(v8bf*)&BldsT[sr][sk]     = b0;
        *(v8bf*)&BldsT[sr][sk + 8] = b1;
#endif
      }
      if (k0 + GT_K < K)
        __builtin_prefetch(&A[(size_t)(brow + sr) * K + k0 + GT_K], 0, 1);
      __syncthreads();

      // fragments (ISA 7.12.2): A lane=row, K = half*8 + j (+16 upper 8)
      v16bf afrag[2], bfrag[4];
#pragma unroll
      for (int i = 0; i < 2; ++i) {
        const __bf16* ar = &Alds[wrow + i * 16 + l15][half * 8];
        afrag[i] = make_frag(ar, ar + 16);
      }
      // B lane=col, K = half*8 + j (+16 upper 8)
#pragma unroll
      for (int j = 0; j < 4; ++j) {
        const __bf16* br = &BldsT[wcol + j * 16 + l15][half * 8];
        bfrag[j] = make_frag(br, br + 16);
      }
#pragma unroll
      for (int i = 0; i < 2; ++i)
#pragma unroll
        for (int j = 0; j < 4; ++j)
          acc[i][j] = WMMA_BF16(afrag[i], bfrag[j], acc[i][j]);
    }
  }

  // epilogue: C layout -> row = r + 8*half, col = l15 per 16x16 sub-tile
#pragma unroll
  for (int i = 0; i < 2; ++i) {
#pragma unroll
    for (int j = 0; j < 4; ++j) {
      const int col = bcol + wcol + j * 16 + l15;
      const float b = bias ? bias[col] : 0.f;
      float g = 1.f, bt = 0.f, mn = 0.f, inv = 1.f;
      if (mode == 1) {
        g = gamma[col]; bt = beta[col]; mn = mean[col];
        inv = rsqrtf(var[col] + 1e-5f);
      }
#pragma unroll
      for (int r = 0; r < 8; ++r) {
        const int row = brow + wrow + i * 16 + r + half * 8;
        float vv = acc[i][j][r] + b;
        if (mode == 1)      vv = elu_f(g * (vv - mn) * inv + bt);
        else if (mode == 2) vv = elu_f(vv);
        else if (mode == 3) vv += residual[(size_t)row * Nc + col];
        if (Cf) Cf[(size_t)row * Nc + col] = vv;
        if (Ch) Ch[(size_t)row * Nc + col] = (__bf16)vv;
      }
    }
  }
}

// ---------------------------------------------------------------------------
// 3) Flash attention over bf16 q/k/v (N x D, head h at cols h*64..h*64+63).
//    grid = (N/128, H); 8 waves; 16 q-rows per wave; 32-key tiles.
// ---------------------------------------------------------------------------
__global__ __launch_bounds__(256) void flash_attn_kernel(
    const __bf16* __restrict__ q, const __bf16* __restrict__ k,
    const __bf16* __restrict__ v, __bf16* __restrict__ o, int N, int D) {
  const int hoff  = blockIdx.y * 64;
  const int qbase = blockIdx.x * 128;
  const int t     = threadIdx.x;
  const int lane  = t & 31;
  const int wv    = t >> 5;
  const int half  = lane >> 4;
  const int l15   = lane & 15;

  __shared__ __bf16 VldsT[64][LDK];    // V tile transposed (d, key)
  __shared__ __bf16 Plds[8][16][LDK];  // wave-private P tiles

  // Q fragments (16 rows x 64 k), 16B global loads, loaded once
  v16bf qf[2];
  {
    const __bf16* qr = &q[(size_t)(qbase + wv * 16 + l15) * D + hoff];
#pragma unroll
    for (int f = 0; f < 2; ++f) {
      const __bf16* p = qr + f * 32 + half * 8;
      qf[f] = make_frag(p, p + 16);
    }
  }

  v8f oacc[4];
#pragma unroll
  for (int c = 0; c < 4; ++c) oacc[c] = (v8f)0.f;
  float mrun[8], lrun[8];
#pragma unroll
  for (int r = 0; r < 8; ++r) { mrun[r] = -3.0e38f; lrun[r] = 0.f; }

  const float scale = 0.125f;  // 1/sqrt(64)

#pragma unroll 1
  for (int kt = 0; kt < N; kt += 32) {
    __syncthreads();
    // stage V tile (32 keys x 64 d) transposed: one 16B load, 8 b16 stores
    {
      const int key = t >> 3;
      const int d8  = (t & 7) * 8;
      v8bf vv = *(const v8bf*)&v[(size_t)(kt + key) * D + hoff + d8];
#pragma unroll
      for (int j = 0; j < 8; ++j) VldsT[d8 + j][key] = vv[j];
    }
    __syncthreads();

    // S = Q * K^T (K rows are d-contiguous in memory = B-fragment order)
    v8f sacc[2];
    sacc[0] = (v8f)0.f; sacc[1] = (v8f)0.f;
#pragma unroll
    for (int f = 0; f < 2; ++f) {
#pragma unroll
      for (int c = 0; c < 2; ++c) {
        const __bf16* kr =
            &k[(size_t)(kt + c * 16 + l15) * D + hoff + f * 32 + half * 8];
        v16bf kfrag = make_frag(kr, kr + 16);
        sacc[c] = WMMA_BF16(qf[f], kfrag, sacc[c]);
      }
    }

    // online softmax: rows in C-frag slot r (row = r + 8*half), cols across
    // the 16 lanes of each half -> shfl_xor butterfly reductions.
    float alpha[8];
#pragma unroll
    for (int r = 0; r < 8; ++r) {
      float s0 = sacc[0][r] * scale;
      float s1 = sacc[1][r] * scale;
      float mx = fmaxf(s0, s1);
#pragma unroll
      for (int m = 8; m >= 1; m >>= 1) mx = fmaxf(mx, __shfl_xor(mx, m, 32));
      float mnew = fmaxf(mrun[r], mx);
      float a = __expf(mrun[r] - mnew);
      float p0 = __expf(s0 - mnew);
      float p1 = __expf(s1 - mnew);
      float rs = p0 + p1;
#pragma unroll
      for (int m = 8; m >= 1; m >>= 1) rs += __shfl_xor(rs, m, 32);
      lrun[r] = lrun[r] * a + rs;
      mrun[r] = mnew;
      alpha[r] = a;
      const int wr = r + half * 8;
      Plds[wv][wr][l15]      = (__bf16)p0;
      Plds[wv][wr][16 + l15] = (__bf16)p1;
    }
#pragma unroll
    for (int c = 0; c < 4; ++c)
#pragma unroll
      for (int r = 0; r < 8; ++r) oacc[c][r] *= alpha[r];

    // same-wave LDS RAW (C-layout spill -> A-layout reload)
    asm volatile("s_wait_dscnt 0" ::: "memory");

    v16bf pfrag;
    {
      const __bf16* pr = &Plds[wv][l15][half * 8];
      pfrag = make_frag(pr, pr + 16);
    }
    // O += P (16x32) * V (32x64)
#pragma unroll
    for (int c = 0; c < 4; ++c) {
      const __bf16* vr = &VldsT[c * 16 + l15][half * 8];
      v16bf vfrag = make_frag(vr, vr + 16);
      oacc[c] = WMMA_BF16(pfrag, vfrag, oacc[c]);
    }
  }

  // finalize: divide by softmax denominator, write bf16
#pragma unroll
  for (int c = 0; c < 4; ++c)
#pragma unroll
    for (int r = 0; r < 8; ++r) {
      const int row = qbase + wv * 16 + r + half * 8;
      const int col = hoff + c * 16 + l15;
      o[(size_t)row * D + col] = (__bf16)(oacc[c][r] / lrun[r]);
    }
}

// ---------------------------------------------------------------------------
// 4) MoE adapter: one wave per node, lane = adapter unit (A = 32 = wave size)
// ---------------------------------------------------------------------------
__global__ __launch_bounds__(256) void adapter_kernel(
    const float* __restrict__ feats, const int* __restrict__ groups,
    const float* __restrict__ AW1, const float* __restrict__ Ab1,
    const float* __restrict__ AW2, const float* __restrict__ Ab2,
    float* __restrict__ out, int N, int F1, int A) {
  const int node = (blockIdx.x * 256 + threadIdx.x) >> 5;
  const int lane = threadIdx.x & 31;
  if (node >= N) return;
  const int g = groups[node];
  const float* W1 = AW1 + (size_t)g * F1 * A;
  const float* fr = feats + (size_t)node * F1;
  float h = 0.f;
#pragma unroll 4
  for (int d = 0; d < F1; ++d) h += fr[d] * W1[(size_t)d * A + lane];
  h += Ab1[g * A + lane];
  h = fmaxf(h, 0.f);
  float ov = h * AW2[(size_t)g * A + lane];
#pragma unroll
  for (int m = 16; m >= 1; m >>= 1) ov += __shfl_xor(ov, m, 32);
  if (lane == 0) out[node] = ov + Ab2[g];
}

// ---------------------------------------------------------------------------
// Orchestration
// ---------------------------------------------------------------------------
extern "C" void kernel_launch(void* const* d_in, const int* in_sizes, int n_in,
                              void* d_out, int out_size, void* d_ws, size_t ws_size,
                              hipStream_t stream) {
  constexpr int N = 4096, D = 256, E = 131072, L = 3, H = 4, F1 = 128, G = 16, A = 32;
  (void)in_sizes; (void)n_in; (void)out_size; (void)ws_size; (void)G;

  const float* x_in   = (const float*)d_in[0];
  const float* ea     = (const float*)d_in[1];
  const float* Wself  = (const float*)d_in[2];
  const float* Wnbr   = (const float*)d_in[3];
  const float* convb  = (const float*)d_in[4];
  const float* bng    = (const float*)d_in[5];
  const float* bnb    = (const float*)d_in[6];
  const float* bnm    = (const float*)d_in[7];
  const float* bnv    = (const float*)d_in[8];
  const float* Wqkv   = (const float*)d_in[9];
  const float* bqkv   = (const float*)d_in[10];
  const float* Wo     = (const float*)d_in[11];
  const float* bo     = (const float*)d_in[12];
  const float* fc1W   = (const float*)d_in[13];
  const float* fc1b   = (const float*)d_in[14];
  const float* AW1    = (const float*)d_in[15];
  const float* Ab1    = (const float*)d_in[16];
  const float* AW2    = (const float*)d_in[17];
  const float* Ab2    = (const float*)d_in[18];
  const int*   eidx   = (const int*)d_in[19];
  const int*   groups = (const int*)d_in[20];
  const int*   esrc   = eidx;
  const int*   edst   = eidx + E;
  float*       out    = (float*)d_out;

  const size_t ND = (size_t)N * D;      // 1,048,576
  const size_t DD = (size_t)D * D;      // 65,536

  // ---- workspace carve-up ----
  float* xa_f  = (float*)d_ws;                  // N*D fp32
  float* xb_f  = xa_f + ND;
  float* agg   = xb_f + ND;
  float* feats = agg + ND;                      // N*F1 fp32
  __bf16* hbase = (__bf16*)(feats + (size_t)N * F1);
  __bf16* xin_h = hbase;                        // bf16 N*D buffers
  __bf16* xa_h  = xin_h + ND;
  __bf16* xb_h  = xa_h + ND;
  __bf16* agg_h = xb_h + ND;
  __bf16* q_h   = agg_h + ND;
  __bf16* k_h   = q_h + ND;
  __bf16* v_h   = k_h + ND;
  __bf16* o_h   = v_h + ND;
  __bf16* xo_h  = o_h + ND;
  __bf16* wselT = xo_h + ND;                    // bf16 weights, transposed
  __bf16* wnbrT = wselT + 3 * DD;
  __bf16* wqkvT = wnbrT + 3 * DD;
  __bf16* woT   = wqkvT + 3 * DD;
  __bf16* fc1T  = woT + DD;                     // F1 x D

  const dim3 blk(256);
  const dim3 tg_dd(D / 32, D / 32);             // (8,8)
  const dim3 tg_f1(D / 32, F1 / 32);            // (8,4)
  const dim3 gg_d(N / GT_M, D / GT_N);          // (32,2)
  const dim3 gg_f1(N / GT_M, F1 / GT_N);        // (32,1)
  const int cvt_blocks  = (int)(ND / 8 / 256);  // 512
  const int scat_blocks = (E * (D / 4) + 255) / 256;

  // ---- one-time weight prep (bf16 + transpose) ----
  for (int i = 0; i < 3; ++i) {
    transpose_w_kernel<<<tg_dd, blk, 0, stream>>>(Wself + i * DD, wselT + i * DD, D, D);
    transpose_w_kernel<<<tg_dd, blk, 0, stream>>>(Wnbr  + i * DD, wnbrT + i * DD, D, D);
    transpose_w_kernel<<<tg_dd, blk, 0, stream>>>(Wqkv  + i * DD, wqkvT + i * DD, D, D);
  }
  transpose_w_kernel<<<tg_dd, blk, 0, stream>>>(Wo, woT, D, D);
  transpose_w_kernel<<<tg_f1, blk, 0, stream>>>(fc1W, fc1T, D, F1);
  cvt_f32_bf16_kernel<<<cvt_blocks, blk, 0, stream>>>(x_in, xin_h, (int)(ND / 8));

  // ---- conv stack ----
  const float*  cur_f = x_in;
  const __bf16* cur_h = xin_h;
  float*  out_f[L] = {xa_f, xb_f, xa_f};
  __bf16* out_h[L] = {xa_h, xb_h, xa_h};
  for (int i = 0; i < L; ++i) {
    (void)hipMemsetAsync(agg, 0, ND * sizeof(float), stream);
    scatter_add_kernel<<<scat_blocks, blk, 0, stream>>>(
        cur_f, esrc, edst, ea, agg, E, D, (i == L - 1) ? 1 : 0);
    cvt_f32_bf16_kernel<<<cvt_blocks, blk, 0, stream>>>(agg, agg_h, (int)(ND / 8));
    const int mode = (i < L - 1) ? 1 : 0;
    gemm_bf16_kernel<<<gg_d, blk, 0, stream>>>(
        cur_h, wselT + i * DD, D, agg_h, wnbrT + i * DD, D,
        convb + (size_t)i * D, out_f[i], out_h[i], N, D, mode,
        (mode == 1) ? bng + (size_t)i * D : nullptr,
        (mode == 1) ? bnb + (size_t)i * D : nullptr,
        (mode == 1) ? bnm + (size_t)i * D : nullptr,
        (mode == 1) ? bnv + (size_t)i * D : nullptr, nullptr);
    cur_f = out_f[i];
    cur_h = out_h[i];
  }
  const float*  xc_f = out_f[L - 1];  // xa_f
  const __bf16* xc_h = out_h[L - 1];  // xa_h

  // ---- attention: QKV (bf16 out only), flash, out-proj(+residual) ----
  __bf16* qkv_out[3] = {q_h, k_h, v_h};
  for (int j = 0; j < 3; ++j)
    gemm_bf16_kernel<<<gg_d, blk, 0, stream>>>(
        xc_h, wqkvT + j * DD, D, nullptr, nullptr, 0,
        bqkv + (size_t)j * D, nullptr, qkv_out[j], N, D, 0,
        nullptr, nullptr, nullptr, nullptr, nullptr);

  flash_attn_kernel<<<dim3(N / 128, H), blk, 0, stream>>>(q_h, k_h, v_h, o_h, N, D);

  gemm_bf16_kernel<<<gg_d, blk, 0, stream>>>(
      o_h, woT, D, nullptr, nullptr, 0, bo, nullptr, xo_h, N, D, 3,
      nullptr, nullptr, nullptr, nullptr, xc_f);

  // ---- FFN (ELU) -> feats (fp32) ----
  gemm_bf16_kernel<<<gg_f1, blk, 0, stream>>>(
      xo_h, fc1T, D, nullptr, nullptr, 0, fc1b, feats, nullptr, N, F1, 2,
      nullptr, nullptr, nullptr, nullptr, nullptr);

  // ---- per-node group adapter -> out ----
  adapter_kernel<<<(N * 32 + 255) / 256, blk, 0, stream>>>(
      feats, groups, AW1, Ab1, AW2, Ab2, out, N, F1, A);
}
